// LSTM_AE_Model_74801150427916
// MI455X (gfx1250) — compile-verified, matrix operand order checked by reading, not run on hardware
//
#include <hip/hip_runtime.h>

typedef __attribute__((ext_vector_type(16))) _Float16 v16h;
typedef __attribute__((ext_vector_type(8)))  _Float16 v8h;
typedef __attribute__((ext_vector_type(8)))  float    v8f;
typedef __attribute__((ext_vector_type(4)))  float    v4f;

#define T_LEN 1024
#define B_N   1024
#define H_N   64
#define STR   72   // LDS hidden-state row stride in halves (16B-aligned chunks, bank-skewed)

// Fast transcendentals: v_exp_f32 + v_rcp_f32 only (TRANS pipe, co-executes with WMMA).
__device__ __forceinline__ float sigmoid_(float v) {
    return __builtin_amdgcn_rcpf(1.0f + __expf(-v));
}
__device__ __forceinline__ float tanh_(float v) {
    v = fminf(fmaxf(v, -15.0f), 15.0f);
    float e = __expf(2.0f * v);
    return 1.0f - 2.0f * __builtin_amdgcn_rcpf(e + 1.0f);
}

// One workgroup = 16 batch rows, 4 waves. Wave w owns hidden units u = 16w..16w+15
// and all four gate tiles (i,f,g,o) for them, so the LSTM cell update is lane-local.
// Recurrent weights live in VGPRs as WMMA B-fragments for the entire sequence.
// Hidden state h ping-pongs through LDS in f16; cell state c stays in registers.
__global__ __launch_bounds__(128)
void lstm_ae_kernel(const float* __restrict__ x,      // [T, B]
                    const float* __restrict__ Wih_e,  // [256, 1]
                    const float* __restrict__ Whh_e,  // [256, 64]
                    const float* __restrict__ bih_e,  // [256]
                    const float* __restrict__ bhh_e,  // [256]
                    const float* __restrict__ Wih_d,  // [4, 64]
                    const float* __restrict__ Whh_d,  // [4, 1]
                    const float* __restrict__ bih_d,  // [4]
                    const float* __restrict__ bhh_d,  // [4]
                    float* __restrict__ out)          // [T, B]
{
    __shared__ __align__(16) _Float16 hbuf[2][16 * STR];

    const int tid  = threadIdx.x;
    const int wave = tid >> 5;        // 0..3 -> hidden-unit group
    const int lane = tid & 31;
    const int n    = lane & 15;       // tile column / A-row index
    const int half = lane >> 4;       // 0/1 lane half
    const int bb   = blockIdx.x * 16; // batch base row
    const int mrow = half * 8;        // first output row this lane accumulates

    // ---- resident recurrent-weight B fragments (f32 -> f16), per gate, per K-half ----
    // gate g column j = 64*g + 16*wave + n ; B(k, j) = Whh_e[j][k] (row j is contiguous)
    v16h bfrag[4][2];
    float wih[4], bias[4];
#pragma unroll
    for (int g = 0; g < 4; ++g) {
        const int j = 64 * g + 16 * wave + n;
        wih[g]  = Wih_e[j];
        bias[g] = bih_e[j] + bhh_e[j];
        const float* wr = Whh_e + j * H_N;
#pragma unroll
        for (int kh = 0; kh < 2; ++kh) {
            v16h f = {};
#pragma unroll
            for (int i = 0; i < 8; ++i) {
                f[i]     = (_Float16)wr[kh * 32 + half * 8 + i];       // k = 32kh + 8*half + i
                f[i + 8] = (_Float16)wr[kh * 32 + 16 + half * 8 + i];  // k = 32kh + 16 + 8*half + i
            }
            bfrag[g][kh] = f;
        }
    }

    // zero initial hidden state (buffer 0)
    for (int idx = tid; idx < 16 * STR; idx += 128) hbuf[0][idx] = (_Float16)0.0f;
    __syncthreads();

    float cstate[8];
#pragma unroll
    for (int r = 0; r < 8; ++r) cstate[r] = 0.0f;

    const float* xptr = x + bb + mrow;
    int p = 0;

#pragma unroll 1
    for (int t = 0; t < T_LEN; ++t) {
        // ---- A fragments (h rows) from LDS: 4x ds_load_b128, documented 16-bit layout ----
        v16h a0 = {}, a1 = {};
        {
            const _Float16* hr = &hbuf[p][n * STR + half * 8];
            v8h q0 = *(const v8h*)(hr);        // k =      8*half .. +7
            v8h q1 = *(const v8h*)(hr + 16);   // k = 16 + 8*half .. +7
            v8h q2 = *(const v8h*)(hr + 32);   // k = 32 + 8*half .. +7
            v8h q3 = *(const v8h*)(hr + 48);   // k = 48 + 8*half .. +7
#pragma unroll
            for (int i = 0; i < 8; ++i) {
                a0[i] = q0[i]; a0[i + 8] = q1[i];
                a1[i] = q2[i]; a1[i + 8] = q3[i];
            }
        }
        // x values for this lane's 8 batch rows (coalesced / broadcast b128 loads)
        v4f xlo = *(const v4f*)(xptr + (size_t)t * B_N);
        v4f xhi = *(const v4f*)(xptr + (size_t)t * B_N + 4);

        // ---- gate pre-activations: bias-seeded C, two chained K32 WMMAs per gate ----
        float gv[4][8];
#pragma unroll
        for (int g = 0; g < 4; ++g) {
            v8f acc;
#pragma unroll
            for (int r = 0; r < 8; ++r) acc[r] = bias[g];
            acc = __builtin_amdgcn_wmma_f32_16x16x32_f16(false, a0, false, bfrag[g][0],
                                                         (short)0, acc, false, false);
            acc = __builtin_amdgcn_wmma_f32_16x16x32_f16(false, a1, false, bfrag[g][1],
                                                         (short)0, acc, false, false);
#pragma unroll
            for (int r = 0; r < 8; ++r) {
                float xv = (r < 4) ? xlo[r] : xhi[r - 4];
                gv[g][r] = acc[r] + xv * wih[g];
            }
        }

        // ---- lane-local LSTM cell update; publish h to the other LDS buffer ----
        _Float16* hw = &hbuf[p ^ 1][0];
#pragma unroll
        for (int r = 0; r < 8; ++r) {
            float ig = sigmoid_(gv[0][r]);
            float fg = sigmoid_(gv[1][r]);
            float gg = tanh_(gv[2][r]);
            float og = sigmoid_(gv[3][r]);
            float c  = fg * cstate[r] + ig * gg;
            cstate[r] = c;
            float h  = og * tanh_(c);
            hw[(mrow + r) * STR + 16 * wave + n] = (_Float16)h;
        }
        p ^= 1;
        __syncthreads();
    }

    // ---- decoder: hidden size 1, constant input h_enc -> hoist input GEMM out of loop ----
    // 16 independent scalar recurrences; lanes 0..15 of wave 0 handle them.
    if (wave == 0 && half == 0) {
        const int m = n;
        const _Float16* he = &hbuf[p][m * STR];
        float gbase[4], wd[4];
#pragma unroll
        for (int q = 0; q < 4; ++q) {
            float s = bih_d[q] + bhh_d[q];
            for (int u = 0; u < H_N; ++u) s += (float)he[u] * Wih_d[q * H_N + u];
            gbase[q] = s;
            wd[q] = Whh_d[q];
        }
        float hd = 0.0f, cd = 0.0f;
        float* op = out + bb + m;
#pragma unroll 1
        for (int t = 0; t < T_LEN; ++t) {
            float i = sigmoid_(gbase[0] + hd * wd[0]);
            float f = sigmoid_(gbase[1] + hd * wd[1]);
            float g = tanh_  (gbase[2] + hd * wd[2]);
            float o = sigmoid_(gbase[3] + hd * wd[3]);
            cd = f * cd + i * g;
            hd = o * tanh_(cd);
            op[(size_t)t * B_N] = hd;   // coalesced across the 16 lanes
        }
    }
}

extern "C" void kernel_launch(void* const* d_in, const int* in_sizes, int n_in,
                              void* d_out, int out_size, void* d_ws, size_t ws_size,
                              hipStream_t stream) {
    const float* x     = (const float*)d_in[0];
    const float* Wih_e = (const float*)d_in[1];
    const float* Whh_e = (const float*)d_in[2];
    const float* bih_e = (const float*)d_in[3];
    const float* bhh_e = (const float*)d_in[4];
    const float* Wih_d = (const float*)d_in[5];
    const float* Whh_d = (const float*)d_in[6];
    const float* bih_d = (const float*)d_in[7];
    const float* bhh_d = (const float*)d_in[8];
    float* out = (float*)d_out;

    dim3 grid(B_N / 16);   // 64 workgroups, one 16-row batch tile each
    dim3 block(128);       // 4 waves: one per 16-hidden-unit group
    hipLaunchKernelGGL(lstm_ae_kernel, grid, block, 0, stream,
                       x, Wih_e, Whh_e, bih_e, bhh_e, Wih_d, Whh_d, bih_d, bhh_d, out);
}
